// HeteroGATLayerBinary_52166672777254
// MI455X (gfx1250) — matrix operand order repaired and checked by reading, not compiled, not run on hardware
//
#include <hip/hip_runtime.h>
#include <hip/hip_bf16.h>
#include <math.h>

#define N_P   50000
#define N_A   50000
#define NEDGE 200000
#define H     4
#define D     32
#define MDIM  32
#define HD    128
#define LRELU_ALPHA 0.2f

typedef _Float16 half_t;
typedef __attribute__((ext_vector_type(16))) _Float16 v16h;
typedef __attribute__((ext_vector_type(8)))  _Float16 v8h;
typedef __attribute__((ext_vector_type(8)))  float    v8f;

// ---------------------------------------------------------------------------
// One-off: convert fp32 activations to f16
// ---------------------------------------------------------------------------
__global__ void f32_to_f16_kernel(const float* __restrict__ in, half_t* __restrict__ out, size_t n) {
  size_t i = (size_t)blockIdx.x * blockDim.x + threadIdx.x;
  if (i < n) out[i] = (half_t)in[i];
}

// ---------------------------------------------------------------------------
// One-off: swizzle weight matrix W[K,Kout] (row-major fp32) into per-lane f16
// B-fragments for v_wmma_f32_16x16x32_f16 SRC1, in this file's contiguous k order (same as A):
//   fragment(tc, kbi): lane L holds halves t=0..15 with K = kb + t + (L>=16)*16,
//   N = tc*16 + (L&15).  Stored as Bp[((tc*nkb + kbi)*32 + L)*16 + t].
// ---------------------------------------------------------------------------
__global__ void pack_B_kernel(const float* __restrict__ W, half_t* __restrict__ Bp,
                              int K, int Kout) {
  const int nkb = K >> 5;
  const int blk = blockIdx.x;                 // blk = tc*nkb + kbi
  const int kb  = (blk % nkb) << 5;
  const int colbase = (blk / nkb) << 4;
  const int lane = threadIdx.x;
  const int hi = lane >> 4, r = lane & 15;
  half_t* out = Bp + ((size_t)blk * 32 + lane) * 16;
#pragma unroll
  for (int t = 0; t < 16; ++t)
    out[t] = (half_t)W[(size_t)(kb + t + (hi << 4)) * Kout + colbase + r];
}

// ---------------------------------------------------------------------------
// WMMA GEMM: C[N,Kout] = Ah[N,K](f16) @ Wpacked(f16) + bias, fp32 accumulate.
// Each wave: 64 rows x 16 cols (4 row-subtiles share each B fragment).
// 8 waves / 256-thread block.
// ---------------------------------------------------------------------------
#define GWAVES 8
#define MT 4
__global__ void gemm_bias_wmma(const half_t* __restrict__ Ah, const half_t* __restrict__ Bp,
                               const float* __restrict__ bias, float* __restrict__ C,
                               int N, int K, int Kout) {
  const int ntc = Kout >> 4;
  const int nkb = K >> 5;
  const int ntr = (N + 63) >> 6;
  const int tile = blockIdx.x * GWAVES + threadIdx.y;
  if (tile >= ntr * ntc) return;              // wave-uniform exit: EXEC all-1 for WMMA
  const int rowbase = (tile / ntc) << 6;
  const int tc      = tile % ntc;
  const int lane = threadIdx.x;               // wave32
  const int r  = lane & 15;
  const int hi = lane >> 4;

  v8f acc[MT] = {};
  for (int kbi = 0; kbi < nkb; ++kbi) {
    const int kb = kbi << 5;
    const v16h b = *(const v16h*)(Bp + (((size_t)tc * nkb + kbi) * 32 + lane) * 16);
#pragma unroll
    for (int m = 0; m < MT; ++m) {
      int arow = rowbase + (m << 4) + r;
      arow = arow < N ? arow : N - 1;         // clamp (masked at store)
      const half_t* Ar = Ah + (size_t)arow * K + kb + (hi << 4);
      const v8h a0 = *(const v8h*)(Ar);       // K = kb + hi*16 + 0..7
      const v8h a1 = *(const v8h*)(Ar + 8);   // K = kb + hi*16 + 8..15
      v16h a;
#pragma unroll
      for (int t = 0; t < 8; ++t) { a[t] = a0[t]; a[8 + t] = a1[t]; }
      acc[m] = __builtin_amdgcn_wmma_f32_16x16x32_f16(false, a, false, b,
                                                      (short)0, acc[m], false, false);
    }
  }
  const int col = (tc << 4) + r;
  const float bv = bias[col];
#pragma unroll
  for (int m = 0; m < MT; ++m) {
#pragma unroll
    for (int rr = 0; rr < 8; ++rr) {
      const int row = rowbase + (m << 4) + rr + (hi << 3);
      if (row < N) C[(size_t)row * Kout + col] = acc[m][rr] + bv;
    }
  }
}

// ---------------------------------------------------------------------------
// Gumbel hard binarize (forward STE value == argmax one-hot, component 1).
// Emits f16 messages directly (GEMM input).
// ---------------------------------------------------------------------------
__global__ void binarize_kernel(const float* __restrict__ Weh, const float* __restrict__ g,
                                const float* __restrict__ Wbin, const float* __restrict__ bbin,
                                half_t* __restrict__ msg, int NM) {
  int i = blockIdx.x * blockDim.x + threadIdx.x;
  if (i >= NM) return;
  float x  = Weh[i];
  float l0 = x * Wbin[0] + bbin[0] + g[2 * i + 0];
  float l1 = x * Wbin[1] + bbin[1] + g[2 * i + 1];
  msg[i] = (l1 > l0) ? (half_t)1.f : (half_t)0.f;   // jnp.argmax ties -> index 0
}

// es/ed[n,h] = dot(F[n,h,:], attn[h,:])
__global__ void node_logits_kernel(const float* __restrict__ F, const float* __restrict__ attn,
                                   float* __restrict__ out, int N) {
  int i = blockIdx.x * blockDim.x + threadIdx.x;
  if (i >= N * H) return;
  int n = i >> 2, h = i & 3;
  const float* f = F + (size_t)n * HD + h * D;
  const float* a = attn + h * D;
  float s = 0.f;
#pragma unroll
  for (int d = 0; d < D; ++d) s += f[d] * a[d];
  out[i] = s;
}

__global__ void init_softmax_kernel(float* __restrict__ mx, float* __restrict__ s, int n) {
  int i = blockIdx.x * blockDim.x + threadIdx.x;
  if (i >= n) return;
  mx[i] = -INFINITY;
  s[i]  = 0.f;
}

__device__ inline void atomicMaxFloat(float* addr, float val) {
  unsigned int* ua = (unsigned int*)addr;
  unsigned int old = __hip_atomic_load(ua, __ATOMIC_RELAXED, __HIP_MEMORY_SCOPE_AGENT), assumed;
  while (__uint_as_float(old) < val) {
    assumed = old;
    old = atomicCAS(ua, assumed, __float_as_uint(val));
    if (old == assumed) break;
  }
}

// pass 1: e = leaky_relu(es[src]+ed[dst]); segment max via atomics
__global__ void edge_logits_kernel(const float* __restrict__ es, const float* __restrict__ ed,
                                   const int* __restrict__ src, const int* __restrict__ dst,
                                   float* __restrict__ e, float* __restrict__ mx, int E) {
  int i = blockIdx.x * blockDim.x + threadIdx.x;
  if (i >= E * H) return;
  int ei = i >> 2, h = i & 3;
  float v = es[src[ei] * H + h] + ed[dst[ei] * H + h];
  v = (v > 0.f) ? v : LRELU_ALPHA * v;
  e[i] = v;
  atomicMaxFloat(&mx[dst[ei] * H + h], v);
}

// pass 2: w = exp(e - mx[dst]); segment sum via atomics (w overwrites e)
__global__ void edge_norm_kernel(float* __restrict__ e, const float* __restrict__ mx,
                                 float* __restrict__ s, const int* __restrict__ dst, int E) {
  int i = blockIdx.x * blockDim.x + threadIdx.x;
  if (i >= E * H) return;
  int ei = i >> 2, h = i & 3;
  float w = __expf(e[i] - mx[dst[ei] * H + h]);
  e[i] = w;
  atomicAdd(&s[dst[ei] * H + h], w);
}

// pass 3: out[dst] += msg[src] * w/s[dst]; one 128-thread block per edge
__global__ void edge_agg_kernel(const float* __restrict__ Msrc, const float* __restrict__ w,
                                const float* __restrict__ s,
                                const int* __restrict__ src, const int* __restrict__ dst,
                                float* __restrict__ out, int E) {
  int ei = blockIdx.x;
  if (ei >= E) return;
  int t = threadIdx.x;           // t = h*32 + d
  int h = t >> 5;
  int sN = src[ei], dN = dst[ei];
  float a = w[ei * H + h] / s[dN * H + h];
  atomicAdd(&out[(size_t)dN * HD + t], Msrc[(size_t)sN * HD + t] * a);
}

__global__ void fill_zero_kernel(float* __restrict__ p, size_t n) {
  size_t i = (size_t)blockIdx.x * blockDim.x + threadIdx.x;
  if (i < n) p[i] = 0.f;
}

__global__ void relu_kernel(float* __restrict__ p, size_t n) {
  size_t i = (size_t)blockIdx.x * blockDim.x + threadIdx.x;
  if (i < n) p[i] = fmaxf(p[i], 0.f);
}

// ---------------------------------------------------------------------------
extern "C" void kernel_launch(void* const* d_in, const int* in_sizes, int n_in,
                              void* d_out, int out_size, void* d_ws, size_t ws_size,
                              hipStream_t stream) {
  (void)in_sizes; (void)n_in; (void)out_size; (void)ws_size;
  const float* feat_P     = (const float*)d_in[0];
  const float* feat_A     = (const float*)d_in[1];
  const float* feat_state = (const float*)d_in[2];
  const float* Wfc_P = (const float*)d_in[3];  const float* bfc_P = (const float*)d_in[4];
  const float* Wfc_A = (const float*)d_in[5];  const float* bfc_A = (const float*)d_in[6];
  const float* Wp2s  = (const float*)d_in[7];  const float* bp2s  = (const float*)d_in[8];
  const float* Wa2s  = (const float*)d_in[9];  const float* ba2s  = (const float*)d_in[10];
  const float* Win   = (const float*)d_in[11]; const float* bin_in= (const float*)d_in[12];
  const float* Wenc_P= (const float*)d_in[13]; const float* benc_P= (const float*)d_in[14];
  const float* Wenc_A= (const float*)d_in[15]; const float* benc_A= (const float*)d_in[16];
  const float* Wdec_P= (const float*)d_in[17]; const float* bdec_P= (const float*)d_in[18];
  const float* Wdec_A= (const float*)d_in[19]; const float* bdec_A= (const float*)d_in[20];
  const float* Wbin  = (const float*)d_in[21]; const float* bbin  = (const float*)d_in[22];
  const float* at_p2p_s=(const float*)d_in[23]; const float* at_p2p_d=(const float*)d_in[24];
  const float* at_p2a_s=(const float*)d_in[25]; const float* at_p2a_d=(const float*)d_in[26];
  const float* at_a2p_s=(const float*)d_in[27]; const float* at_a2p_d=(const float*)d_in[28];
  const float* at_a2a_s=(const float*)d_in[29]; const float* at_a2a_d=(const float*)d_in[30];
  const float* at_p2s_s=(const float*)d_in[31]; const float* at_p2s_d=(const float*)d_in[32];
  const float* at_a2s_s=(const float*)d_in[33]; const float* at_a2s_d=(const float*)d_in[34];
  const float* gumbel_P=(const float*)d_in[35]; const float* gumbel_A=(const float*)d_in[36];
  const int* src_p2p=(const int*)d_in[37]; const int* dst_p2p=(const int*)d_in[38];
  const int* src_p2a=(const int*)d_in[39]; const int* dst_p2a=(const int*)d_in[40];
  const int* src_a2p=(const int*)d_in[41]; const int* dst_a2p=(const int*)d_in[42];
  const int* src_a2a=(const int*)d_in[43]; const int* dst_a2a=(const int*)d_in[44];
  const int* src_p2s=(const int*)d_in[45]; const int* dst_p2s=(const int*)d_in[46];
  const int* src_a2s=(const int*)d_in[47]; const int* dst_a2s=(const int*)d_in[48];

  // ---- workspace layout (float units, each chunk 64B-aligned) ----
  float* ws = (float*)d_ws;
  size_t off = 0;
  auto alloc_f = [&](size_t nfloats) { float* p = ws + off; off += (nfloats + 15) & ~(size_t)15; return p; };
  auto alloc_h = [&](size_t nhalves) { return (half_t*)alloc_f((nhalves + 1) / 2); };

  float* Whp   = alloc_f((size_t)N_P * HD);
  float* Wha   = alloc_f((size_t)N_A * HD);
  float* Whp2s = alloc_f((size_t)N_P * HD);
  float* Wha2s = alloc_f((size_t)N_A * HD);
  float* Whin  = alloc_f(HD);
  float* encP  = alloc_f((size_t)N_P * MDIM);
  float* encA  = alloc_f((size_t)N_A * MDIM);
  float* Mbuf  = alloc_f((size_t)N_P * HD);     // decoded messages (reused per edge type)
  float* esb   = alloc_f((size_t)N_P * H);
  float* edb   = alloc_f((size_t)N_P * H);
  float* ebuf  = alloc_f((size_t)NEDGE * H);
  float* mxb   = alloc_f((size_t)N_P * H);
  float* sbuf  = alloc_f((size_t)N_P * H);
  // f16 activations
  half_t* featP_h = alloc_h((size_t)N_P * 64);
  half_t* featA_h = alloc_h((size_t)N_A * 64);
  half_t* featS_h = alloc_h(64);
  half_t* msgP_h  = alloc_h((size_t)N_P * MDIM);
  half_t* msgA_h  = alloc_h((size_t)N_A * MDIM);
  // f16 pre-swizzled weights (size == K*Kout halves each)
  half_t* Bp_fcP = alloc_h(64 * HD);   half_t* Bp_fcA = alloc_h(64 * HD);
  half_t* Bp_p2s = alloc_h(64 * HD);   half_t* Bp_a2s = alloc_h(64 * HD);
  half_t* Bp_in  = alloc_h(64 * HD);
  half_t* Bp_encP = alloc_h(64 * MDIM); half_t* Bp_encA = alloc_h(64 * MDIM);
  half_t* Bp_decP = alloc_h(MDIM * HD); half_t* Bp_decA = alloc_h(MDIM * HD);

  float* outP = (float*)d_out;
  float* outA = outP + (size_t)N_P * HD;
  float* outS = outA + (size_t)N_A * HD;
  const size_t out_total = (size_t)N_P * HD + (size_t)N_A * HD + HD;

  auto cdiv = [](long long a, long long b) { return (int)((a + b - 1) / b); };

  auto packW = [&](const float* W, half_t* Bp, int K, int Kout) {
    pack_B_kernel<<<(K >> 5) * (Kout >> 4), 32, 0, stream>>>(W, Bp, K, Kout);
  };
  auto gemm = [&](const half_t* Ah, const half_t* Bp, const float* b, float* C,
                  int N, int K, int Kout) {
    int tiles = ((N + 63) / 64) * (Kout / 16);
    gemm_bias_wmma<<<cdiv(tiles, GWAVES), dim3(32, GWAVES), 0, stream>>>(Ah, Bp, b, C, N, K, Kout);
  };
  auto run_gat = [&](const float* Msrc, int n_src, const float* Whdst, int n_dst,
                     const float* asrc, const float* adst,
                     const int* src, const int* dst, int nE, float* outbuf) {
    node_logits_kernel<<<cdiv((long long)n_src * H, 256), 256, 0, stream>>>(Msrc, asrc, esb, n_src);
    node_logits_kernel<<<cdiv((long long)n_dst * H, 256), 256, 0, stream>>>(Whdst, adst, edb, n_dst);
    init_softmax_kernel<<<cdiv((long long)n_dst * H, 256), 256, 0, stream>>>(mxb, sbuf, n_dst * H);
    edge_logits_kernel<<<cdiv((long long)nE * H, 256), 256, 0, stream>>>(esb, edb, src, dst, ebuf, mxb, nE);
    edge_norm_kernel<<<cdiv((long long)nE * H, 256), 256, 0, stream>>>(ebuf, mxb, sbuf, dst, nE);
    edge_agg_kernel<<<nE, HD, 0, stream>>>(Msrc, ebuf, sbuf, src, dst, outbuf, nE);
  };

  // 0) zero output accumulators; convert activations; swizzle weights
  fill_zero_kernel<<<cdiv((long long)out_total, 256), 256, 0, stream>>>((float*)d_out, out_total);
  f32_to_f16_kernel<<<cdiv((long long)N_P * 64, 256), 256, 0, stream>>>(feat_P, featP_h, (size_t)N_P * 64);
  f32_to_f16_kernel<<<cdiv((long long)N_A * 64, 256), 256, 0, stream>>>(feat_A, featA_h, (size_t)N_A * 64);
  f32_to_f16_kernel<<<1, 64, 0, stream>>>(feat_state, featS_h, 64);
  packW(Wfc_P, Bp_fcP, 64, HD);   packW(Wfc_A, Bp_fcA, 64, HD);
  packW(Wp2s,  Bp_p2s, 64, HD);   packW(Wa2s,  Bp_a2s, 64, HD);
  packW(Win,   Bp_in,  64, HD);
  packW(Wenc_P, Bp_encP, 64, MDIM); packW(Wenc_A, Bp_encA, 64, MDIM);
  packW(Wdec_P, Bp_decP, MDIM, HD); packW(Wdec_A, Bp_decA, MDIM, HD);

  // 1) per-type feature transforms (WMMA)
  gemm(featP_h, Bp_fcP, bfc_P, Whp,   N_P, 64, HD);
  gemm(featA_h, Bp_fcA, bfc_A, Wha,   N_A, 64, HD);
  gemm(featP_h, Bp_p2s, bp2s,  Whp2s, N_P, 64, HD);
  gemm(featA_h, Bp_a2s, ba2s,  Wha2s, N_A, 64, HD);
  gemm(featS_h, Bp_in,  bin_in, Whin, 1,   64, HD);

  // 2) encoder + gumbel hard binarize (f16 messages)
  gemm(featP_h, Bp_encP, benc_P, encP, N_P, 64, MDIM);
  gemm(featA_h, Bp_encA, benc_A, encA, N_A, 64, MDIM);
  binarize_kernel<<<cdiv((long long)N_P * MDIM, 256), 256, 0, stream>>>(encP, gumbel_P, Wbin, bbin, msgP_h, N_P * MDIM);
  binarize_kernel<<<cdiv((long long)N_A * MDIM, 256), 256, 0, stream>>>(encA, gumbel_A, Wbin, bbin, msgA_h, N_A * MDIM);

  // 3) edge types (decoder WMMA GEMM + GAT); p2p & a2p share outP accum, etc.
  gemm(msgP_h, Bp_decP, bdec_P, Mbuf, N_P, MDIM, HD);
  run_gat(Mbuf, N_P, Whp, N_P, at_p2p_s, at_p2p_d, src_p2p, dst_p2p, NEDGE, outP);
  gemm(msgA_h, Bp_decP, bdec_P, Mbuf, N_A, MDIM, HD);
  run_gat(Mbuf, N_A, Whp, N_P, at_a2p_s, at_a2p_d, src_a2p, dst_a2p, NEDGE, outP);
  gemm(msgP_h, Bp_decA, bdec_A, Mbuf, N_P, MDIM, HD);
  run_gat(Mbuf, N_P, Wha, N_A, at_p2a_s, at_p2a_d, src_p2a, dst_p2a, NEDGE, outA);
  gemm(msgA_h, Bp_decA, bdec_A, Mbuf, N_A, MDIM, HD);
  run_gat(Mbuf, N_A, Wha, N_A, at_a2a_s, at_a2a_d, src_a2a, dst_a2a, NEDGE, outA);
  run_gat(Whp2s, N_P, Whin, 1, at_p2s_s, at_p2s_d, src_p2s, dst_p2s, N_P, outS);
  run_gat(Wha2s, N_A, Whin, 1, at_a2s_s, at_a2s_d, src_a2s, dst_a2s, N_A, outS);

  // 4) final ReLU in place
  relu_kernel<<<cdiv((long long)out_total, 256), 256, 0, stream>>>((float*)d_out, out_total);
}
